// SelfAttention_83743272337485
// MI455X (gfx1250) — matrix operand order repaired and checked by reading, not compiled, not run on hardware
//
#include <hip/hip_runtime.h>
#include <math.h>

// Problem constants (from reference): B=4, L=2048, D=1024, H=64
#define BB 4
#define LL 2048
#define DD 1024
#define HH 64

// gfx1250 async global->LDS path (ASYNCcnt-tracked), guarded so a missing
// builtin falls back to the register staging path.
#if defined(__has_builtin)
#if __has_builtin(__builtin_amdgcn_global_load_async_to_lds_b128) && \
    __has_builtin(__builtin_amdgcn_s_wait_asynccnt)
#define HAVE_ASYNC_LDS 1
#endif
#endif
#ifndef HAVE_ASYNC_LDS
#define HAVE_ASYNC_LDS 0
#endif

#if HAVE_ASYNC_LDS
// ROCm 7.2 declares the builtin as (v4i addrspace(1)*, v4i addrspace(3)*, Ii, Ii)
// per the compiler diagnostic (addrspace(1) prints as __device__ in HIP mode).
typedef int v4i32 __attribute__((vector_size(16)));
typedef __attribute__((address_space(1))) v4i32 gv4i32;  // global b128 chunk
typedef __attribute__((address_space(3))) v4i32 sv4i32;  // LDS b128 chunk
#endif

typedef __attribute__((ext_vector_type(16))) __bf16 bf16x16;
typedef __attribute__((ext_vector_type(8)))  __bf16 bf16x8;
typedef __attribute__((ext_vector_type(8)))  float  f32x8;

static __device__ __forceinline__ f32x8 wmma_bf16(bf16x16 a, bf16x16 b, f32x8 c) {
  // D = A(16x32 bf16) * B(32x16 bf16) + C(16x16 f32)
  return __builtin_amdgcn_wmma_f32_16x16x32_bf16(false, a, false, b, (short)0, c,
                                                 false, false);
}

static __device__ __forceinline__ bf16x16 pack16(bf16x8 lo, bf16x8 hi) {
  return __builtin_shufflevector(lo, hi, 0, 1, 2, 3, 4, 5, 6, 7, 8, 9, 10, 11, 12,
                                 13, 14, 15);
}

static __device__ __forceinline__ bf16x16 cvt_pack(f32x8 lo, f32x8 hi) {
  return pack16(__builtin_convertvector(lo, bf16x8),
                __builtin_convertvector(hi, bf16x8));
}

// ---------------------------------------------------------------------------
// Kernel 1: Q = x @ Wq^T, output bf16 [B*L, H] row-major into workspace.
// One wave per 16-row tile of x; K-loop over D in 32-wide bf16 WMMA steps.
// A-fragment (x rows): lane half h holds K 16h..16h+15 contiguous, same order as B.
// B-fragment (Wq rows as columns): lane=output col, elements K contiguous per half.
// ---------------------------------------------------------------------------
__global__ __launch_bounds__(256) void qproj_wmma(const float* __restrict__ x,
                                                  const float* __restrict__ Wq,
                                                  __bf16* __restrict__ Qb) {
  const int lane = threadIdx.x & 31;
  const int wave = threadIdx.x >> 5;
  const int n    = lane & 15;
  const int half = lane >> 4;
  const int hb8  = half * 8;   // D-fragment row base for this lane half
  const int h16  = half * 16;  // B-fragment K-base for this lane half

  const int row0 = (blockIdx.x * 8 + wave) * 16;  // 512 tiles over B*L=8192 rows
  const float* xrow = x + (size_t)(row0 + n) * DD;

  f32x8 zc = {};
  f32x8 acc[4];
#pragma unroll
  for (int t = 0; t < 4; ++t) acc[t] = zc;

  for (int kb = 0; kb < DD; kb += 32) {
    bf16x16 afrag = cvt_pack(*(const f32x8*)(xrow + kb + h16),
                             *(const f32x8*)(xrow + kb + h16 + 8));
#pragma unroll
    for (int t = 0; t < 4; ++t) {
      const float* wrow = Wq + (size_t)(t * 16 + n) * DD + kb + h16;
      bf16x16 bfrag = cvt_pack(*(const f32x8*)(wrow), *(const f32x8*)(wrow + 8));
      acc[t] = wmma_bf16(afrag, bfrag, acc[t]);
    }
  }
  // D-fragment: lane = col n, VGPR r -> row r + 8*half
#pragma unroll
  for (int t = 0; t < 4; ++t)
#pragma unroll
    for (int r = 0; r < 8; ++r)
      Qb[(size_t)(row0 + r + hb8) * HH + t * 16 + n] = (__bf16)acc[t][r];
}

// ---------------------------------------------------------------------------
// Kernel 2: flash attention, q=k=v=Q (reference bug preserved), causal,
// scale = 1/sqrt(D)=1/32. 8 waves/block; wave w owns queries [l0, l0+16).
// Workgroup-shared 32-key blocks staged in LDS (both layouts); the (key,h)
// copy goes through the gfx1250 async global->LDS DMA path when available.
// S^T = K*Q^T so score fragment == P A-operand layout (lane-local softmax->P).
// ---------------------------------------------------------------------------
__global__ __launch_bounds__(256) void flash_attn_wmma(const __bf16* __restrict__ Qb,
                                                       float* __restrict__ out) {
  __shared__ __align__(32) __bf16 Kt[32][72];  // (key, h), padded stride 144B
  __shared__ __align__(32) __bf16 Vt[64][48];  // (h, key), padded stride 96B

  const int tid  = threadIdx.x;
  const int lane = tid & 31;
  const int wave = tid >> 5;
  const int n    = lane & 15;
  const int half = lane >> 4;
  const int hb8  = half * 8;
  const int h16  = half * 16;

  const int wgPerB = LL / 128;  // 16
  const int b      = blockIdx.x / wgPerB;
  const int wg_l0  = (blockIdx.x % wgPerB) * 128;
  const int l0     = wg_l0 + wave * 16;
  const int q      = l0 + n;  // this lane's query row (score fragment column)

  const __bf16* Qbat = Qb + (size_t)b * LL * HH;

  // Q-tile B-fragments for S^T (loop-invariant): lane = query col, K = h.
  bf16x16 qfrag[2];
#pragma unroll
  for (int hb = 0; hb < 2; ++hb)
    qfrag[hb] = *(const bf16x16*)(Qbat + (size_t)(l0 + n) * HH + hb * 32 + h16);

  f32x8 zc = {};
  f32x8 acc[4];
#pragma unroll
  for (int t = 0; t < 4; ++t) acc[t] = zc;
  float run_max = -INFINITY, run_sum = 0.0f;

  const int nBlk = wg_l0 / 32 + 4;  // keys 0 .. wg_l0+127
  const int krow = tid >> 3;        // 0..31 (key row this thread stages)
  const int hcol = (tid & 7) * 8;   // 0..56 (8 h-values)

  for (int blk = 0; blk < nBlk; ++blk) {
    const int m0 = blk * 32;
    {  // cooperative stage of 32x64 bf16 key tile, normal + transposed
      const __bf16* gsrc = Qbat + (size_t)(m0 + krow) * HH + hcol;
#if HAVE_ASYNC_LDS
      // Async DMA of the (key,h) copy straight into LDS (no VGPR round-trip):
      // emits global_load_async_to_lds_b128, tracked by ASYNCcnt.
      __builtin_amdgcn_global_load_async_to_lds_b128(
          (gv4i32*)(void*)gsrc, (sv4i32*)(void*)&Kt[krow][hcol], 0, 0);
#endif
      // Register copy of the same cacheline feeds the transposed (h,key) tile
      // (second read of the line is L0-resident).
      bf16x8 v = *(const bf16x8*)gsrc;
#if !HAVE_ASYNC_LDS
      *(bf16x8*)&Kt[krow][hcol] = v;
#endif
#pragma unroll
      for (int e = 0; e < 8; ++e) Vt[hcol + e][krow] = v[e];
      if (blk + 1 < nBlk)  // gfx1250 global_prefetch_b8 for next tile
        __builtin_prefetch(Qbat + (size_t)(m0 + 32 + krow) * HH + hcol, 0, 1);
#if HAVE_ASYNC_LDS
      __builtin_amdgcn_s_wait_asynccnt(0);  // drain DMA before the barrier
#endif
    }
    __syncthreads();

    if (m0 <= l0 + 15) {  // causal: this wave still has live keys in the block
      // --- S^T = K * Q^T : c0 = keys m0..m0+15, c1 = keys m0+16..m0+31 ---
      f32x8 c0 = zc, c1 = zc;
#pragma unroll
      for (int hb = 0; hb < 2; ++hb) {
        const int ho = hb * 32 + h16;
        bf16x16 a0 = pack16(*(const bf16x8*)&Kt[n][ho],
                            *(const bf16x8*)&Kt[n][ho + 8]);
        bf16x16 a1 = pack16(*(const bf16x8*)&Kt[16 + n][ho],
                            *(const bf16x8*)&Kt[16 + n][ho + 8]);
        c0 = wmma_bf16(a0, qfrag[hb], c0);
        c1 = wmma_bf16(a1, qfrag[hb], c1);
      }
      // --- online softmax (fp32 state; lane pair L,L^16 shares one query) ---
      const float scale = 0.03125f;  // 1/sqrt(1024)
      float s0[8], s1[8], mloc = -INFINITY;
#pragma unroll
      for (int r = 0; r < 8; ++r) {
        s0[r] = (m0 + r + hb8 <= q) ? c0[r] * scale : -INFINITY;
        s1[r] = (m0 + 16 + r + hb8 <= q) ? c1[r] * scale : -INFINITY;
        mloc  = fmaxf(mloc, fmaxf(s0[r], s1[r]));
      }
      const float mblk  = fmaxf(mloc, __shfl_xor(mloc, 16, 32));
      const float nmax  = fmaxf(run_max, mblk);
      const float alpha = __expf(run_max - nmax);
      run_max           = nmax;

      bf16x16 pf;  // P as A-operand: elements {0-7}=c0 rows, {8-15}=c1 rows
      float ls = 0.0f;
#pragma unroll
      for (int r = 0; r < 8; ++r) {
        const float e0 = __expf(s0[r] - nmax);
        const float e1 = __expf(s1[r] - nmax);
        ls += e0 + e1;
        pf[r]     = (__bf16)e0;
        pf[r + 8] = (__bf16)e1;
      }
      run_sum = run_sum * alpha + (ls + __shfl_xor(ls, 16, 32));

      // broadcast per-query rescale to O-accumulator rows (row m = r + 8*half)
      float arow[8];
#pragma unroll
      for (int r = 0; r < 8; ++r) arow[r] = __shfl(alpha, r + hb8, 32);

      // --- O += P * V over this 32-key block, N = h in 4 tiles of 16 ---
#pragma unroll
      for (int t = 0; t < 4; ++t) {
        bf16x16 vfrag = pack16(*(const bf16x8*)&Vt[t * 16 + n][hb8],
                               *(const bf16x8*)&Vt[t * 16 + n][hb8 + 16]);
        f32x8 c = acc[t];
#pragma unroll
        for (int r = 0; r < 8; ++r) c[r] *= arow[r];
        acc[t] = wmma_bf16(pf, vfrag, c);
      }
    }
    __syncthreads();
  }

  float inv[8];
#pragma unroll
  for (int r = 0; r < 8; ++r) inv[r] = 1.0f / __shfl(run_sum, r + hb8, 32);
#pragma unroll
  for (int t = 0; t < 4; ++t)
#pragma unroll
    for (int r = 0; r < 8; ++r)
      out[((size_t)b * LL + l0 + r + hb8) * HH + t * 16 + n] = acc[t][r] * inv[r];
}

// ---------------------------------------------------------------------------
extern "C" void kernel_launch(void* const* d_in, const int* in_sizes, int n_in,
                              void* d_out, int out_size, void* d_ws, size_t ws_size,
                              hipStream_t stream) {
  const float* x  = (const float*)d_in[0];   // [B, L, D] f32
  const float* Wq = (const float*)d_in[1];   // [H, D]    f32
  float* out      = (float*)d_out;           // [B, L, H] f32
  __bf16* Qb      = (__bf16*)d_ws;           // [B*L, H] bf16 (1 MB scratch)

  // 512 row-tiles of 16 over B*L=8192; 8 waves per 256-thread block.
  qproj_wmma<<<64, 256, 0, stream>>>(x, Wq, Qb);
  // B * (L/128) = 64 workgroups of 256 threads (8 query tiles each).
  flash_attn_wmma<<<64, 256, 0, stream>>>(Qb, out);
}
